// Attention_89395449299117
// MI455X (gfx1250) — matrix operand order repaired and checked by reading, not compiled, not run on hardware
//
#include <hip/hip_runtime.h>

// CDNA5 / gfx1250 flash-attention forward.
// f16 WMMA (f32 accum) + Tensor Data Mover double-buffered K/V staging.
// B=2, H=16, S=2048, D=64, causal. Inputs/outputs fp32.

typedef __attribute__((ext_vector_type(16))) _Float16 v16h;
typedef __attribute__((ext_vector_type(8)))  _Float16 v8h;
typedef __attribute__((ext_vector_type(8)))  float    v8f;
typedef __attribute__((ext_vector_type(4)))  unsigned int v4u;
typedef __attribute__((ext_vector_type(8)))  unsigned int v8u;

#define WMMA_F16(A, B, C) \
  __builtin_amdgcn_wmma_f32_16x16x32_f16(false, (A), false, (B), (short)0, (C), false, false)

namespace {
constexpr int S_LEN = 2048;
constexpr int D_DIM = 64;
constexpr int QBLK  = 64;   // query rows per workgroup
constexpr int KT    = 32;   // key tile
constexpr float L2E = 1.44269504088896340736f;
}

// Issue one TDM 2-D tile load: 32 rows x 64 f32 from global -> LDS.
// Short (2-operand) form: descriptor groups 2/3 are NULL (<=2D tensor).
// D# group0: count=1 | lds_addr | global_addr[56:0] | type=2
// D# group1: data_size=4B, tensor_dim0=64, tensor_dim1=32,
//            tile_dim0=64, tile_dim1=32, tensor_dim0_stride=64
__device__ __forceinline__ void tdm_load_tile_2d(const float* gptr, unsigned lds_off) {
    const unsigned long long ga = (unsigned long long)(uintptr_t)gptr;
    v4u g0;
    g0[0] = 1u;                                                   // count=1 (valid)
    g0[1] = lds_off;                                              // LDS byte address
    g0[2] = (unsigned)ga;                                         // global_addr[31:0]
    g0[3] = ((unsigned)(ga >> 32) & 0x01FFFFFFu) | (2u << 30);    // addr[56:32] | type=2
    v8u g1;
    g1[0] = (2u << 16);        // workgroup_mask=0, data_size=2 (4 bytes)
    g1[1] = (64u << 16);       // tensor_dim0[15:0]=64 (bits 63:48)
    g1[2] = (32u << 16);       // tensor_dim0 hi=0 | tensor_dim1[15:0]=32
    g1[3] = (64u << 16);       // tensor_dim1 hi=0 | tile_dim0=64
    g1[4] = 32u;               // tile_dim1=32, tile_dim2=0
    g1[5] = 64u;               // tensor_dim0_stride[31:0]=64
    g1[6] = 0u;                // stride0 hi | stride1 lo
    g1[7] = 0u;                // stride1 hi
    asm volatile("tensor_load_to_lds %0, %1" :: "s"(g0), "s"(g1) : "memory");
}

__device__ __forceinline__ unsigned lds_offset_of(const void* p) {
    // Generic-pointer LDS aperture: addr[31:0] is the LDS byte offset.
    return (unsigned)(uintptr_t)p;
}

__global__ __launch_bounds__(128)
void fa_fwd_f16wmma(const float* __restrict__ Q, const float* __restrict__ K,
                    const float* __restrict__ V, float* __restrict__ O)
{
    // Double-buffered raw f32 staging tiles (written by TDM).
    __shared__ alignas(16) float   stgK[2][KT * D_DIM];   // [key][d] f32
    __shared__ alignas(16) float   stgV[2][KT * D_DIM];   // [key][d] f32
    // WMMA-ready f16 tiles.
    __shared__ alignas(32) _Float16 sK[KT * D_DIM];       // [key][d]
    __shared__ alignas(32) _Float16 sV[D_DIM * KT];       // [d][key] (transposed)
    __shared__ alignas(32) _Float16 sP[4][16 * KT];       // per-wave P [q][k]

    const int lane  = threadIdx.x & 31;
    const int wid   = threadIdx.x >> 5;
    const int ln16  = lane & 15;
    const int hi    = lane >> 4;        // wave half
    const int rbase = hi * 8;           // C-layout row offset of this half

    const int bh    = blockIdx.y;       // fused batch*head
    const int qbase = blockIdx.x * QBLK;
    const int qw    = qbase + wid * 16; // this wave's first query row
    const size_t base = (size_t)bh * S_LEN * D_DIM;

    const float NEG_INF = -__builtin_inff();

    // ---- Q -> A-fragments (16x32 f16 layout), scale 1/sqrt(D)=0.125 folded in ----
    v16h aQ[2];
    {
        const float* qrow = Q + base + (size_t)(qw + ln16) * D_DIM;
        #pragma unroll
        for (int c = 0; c < 2; ++c) {
            const int off0 = c * 32 + (hi ? 16 : 0);
            #pragma unroll
            for (int i = 0; i < 8; ++i) {
                aQ[c][i]     = (_Float16)(qrow[off0 + i]      * 0.125f);
                aQ[c][8 + i] = (_Float16)(qrow[off0 + 8 + i] * 0.125f);
            }
        }
    }

    // ---- running softmax state + output accumulators ----
    float mrow[8], lrow[8];
    #pragma unroll
    for (int r = 0; r < 8; ++r) { mrow[r] = NEG_INF; lrow[r] = 0.0f; }
    v8f acc[4] = {v8f{}, v8f{}, v8f{}, v8f{}};

    const int ktiles = (qbase + QBLK) / KT;   // causal upper bound for this WG
    const int qmax   = qw + 15;

    // ---- prologue: TDM-prefetch tile 0 into buffer 0 (wave 0 only) ----
    if (wid == 0) {
        tdm_load_tile_2d(K + base, lds_offset_of(&stgK[0][0]));
        tdm_load_tile_2d(V + base, lds_offset_of(&stgV[0][0]));
    }

    for (int tile = 0; tile < ktiles; ++tile) {
        const int kb  = tile * KT;
        const int buf = tile & 1;

        if (wid == 0) {
            if (tile + 1 < ktiles) {   // TDM-prefetch next tile into the other buffer
                const size_t koff = base + (size_t)(kb + KT) * D_DIM;
                tdm_load_tile_2d(K + koff, lds_offset_of(&stgK[buf ^ 1][0]));
                tdm_load_tile_2d(V + koff, lds_offset_of(&stgV[buf ^ 1][0]));
                __builtin_amdgcn_s_wait_tensorcnt((short)2);  // tile `tile` complete
            } else {
                __builtin_amdgcn_s_wait_tensorcnt((short)0);
            }
        }
        __syncthreads();  // staged f32 visible; previous tile's compute done

        // ---- conversion pass: staged f32 -> WMMA-layout f16 (128 threads) ----
        {
            const int key  = threadIdx.x >> 2;        // 0..31
            const int dseg = (threadIdx.x & 3) * 16;  // 0,16,32,48
            const float* kf = &stgK[buf][key * D_DIM + dseg];
            const float* vf = &stgV[buf][key * D_DIM + dseg];
            v16h tk;
            _Float16 tv[16];
            #pragma unroll
            for (int i = 0; i < 16; ++i) {
                tk[i] = (_Float16)kf[i];
                tv[i] = (_Float16)vf[i];
            }
            *(v16h*)&sK[key * D_DIM + dseg] = tk;     // [key][d]
            #pragma unroll
            for (int i = 0; i < 16; ++i)              // transpose into [d][key]
                sV[(dseg + i) * KT + key] = tv[i];
        }
        __syncthreads();  // f16 tiles ready

        if (kb <= qmax) {   // wave-uniform: skip tiles fully above the diagonal
            // ---- S = Q K^T : two 16x16 score tiles, contraction D=64 in 2 chunks ----
            v8f s[2] = {v8f{}, v8f{}};
            #pragma unroll
            for (int t = 0; t < 2; ++t) {
                #pragma unroll
                for (int c = 0; c < 2; ++c) {
                    const v16h bK = *(const v16h*)&sK[(t * 16 + ln16) * D_DIM + c * 32 + hi * 16];
                    s[t] = WMMA_F16(aQ[c], bK, s[t]);
                }
            }

            // ---- mask + online softmax (rows live per-VGPR across 16-lane halves) ----
            const bool needMask = (kb + KT - 1) > qw;
            float p0v[8], p1v[8];
            #pragma unroll
            for (int r = 0; r < 8; ++r) {
                float a0 = s[0][r], a1 = s[1][r];
                if (needMask) {
                    const int qg = qw + rbase + r;
                    if (kb + ln16      > qg) a0 = NEG_INF;
                    if (kb + 16 + ln16 > qg) a1 = NEG_INF;
                }
                float mr = fmaxf(a0, a1);
                mr = fmaxf(mr, __shfl_xor(mr, 1, 32));
                mr = fmaxf(mr, __shfl_xor(mr, 2, 32));
                mr = fmaxf(mr, __shfl_xor(mr, 4, 32));
                mr = fmaxf(mr, __shfl_xor(mr, 8, 32));
                const float mnew  = fmaxf(mrow[r], mr);
                const float alpha = __builtin_exp2f((mrow[r] - mnew) * L2E);
                mrow[r] = mnew;
                const float p0 = __builtin_exp2f((a0 - mnew) * L2E);
                const float p1 = __builtin_exp2f((a1 - mnew) * L2E);
                float rs = p0 + p1;
                rs += __shfl_xor(rs, 1, 32);
                rs += __shfl_xor(rs, 2, 32);
                rs += __shfl_xor(rs, 4, 32);
                rs += __shfl_xor(rs, 8, 32);
                lrow[r] = lrow[r] * alpha + rs;
                #pragma unroll
                for (int g = 0; g < 4; ++g) acc[g][r] *= alpha;
                p0v[r] = p0; p1v[r] = p1;
            }

            // ---- P: C-layout -> A-layout via wave-private LDS transpose ----
            _Float16* pw = sP[wid];
            #pragma unroll
            for (int r = 0; r < 8; ++r) {
                pw[(rbase + r) * KT + ln16]      = (_Float16)p0v[r];
                pw[(rbase + r) * KT + 16 + ln16] = (_Float16)p1v[r];
            }
            asm volatile("s_wait_dscnt 0x0" ::: "memory");
            v16h aP;
            {
                const v8h lo = *(const v8h*)&pw[ln16 * KT + (hi ? 16 : 0)];
                const v8h hh = *(const v8h*)&pw[ln16 * KT + 8 + (hi ? 16 : 0)];
                #pragma unroll
                for (int i = 0; i < 8; ++i) { aP[i] = lo[i]; aP[8 + i] = hh[i]; }
            }

            // ---- O += P V : 4 slabs of 16 output columns ----
            #pragma unroll
            for (int g = 0; g < 4; ++g) {
                const v16h bV = *(const v16h*)&sV[(g * 16 + ln16) * KT + hi * 16];
                acc[g] = WMMA_F16(aP, bV, acc[g]);
            }
        }
    }

    // ---- normalize and store (fp32) ----
    float* orow = O + base;
    #pragma unroll
    for (int r = 0; r < 8; ++r) {
        const int qg = qw + rbase + r;
        const float inv = 1.0f / lrow[r];
        #pragma unroll
        for (int g = 0; g < 4; ++g)
            orow[(size_t)qg * D_DIM + g * 16 + ln16] = acc[g][r] * inv;
    }
}

extern "C" void kernel_launch(void* const* d_in, const int* in_sizes, int n_in,
                              void* d_out, int out_size, void* d_ws, size_t ws_size,
                              hipStream_t stream) {
    (void)in_sizes; (void)n_in; (void)d_ws; (void)ws_size; (void)out_size;
    const float* Q = (const float*)d_in[0];
    const float* K = (const float*)d_in[1];
    const float* V = (const float*)d_in[2];
    // d_in[3] is the causal mask; causality is applied analytically in-kernel.
    float* O = (float*)d_out;

    dim3 grid(S_LEN / QBLK, 2 * 16);   // (query blocks, B*H)
    dim3 block(128);                   // 4 waves of 32
    fa_fwd_f16wmma<<<grid, block, 0, stream>>>(Q, K, V, O);
}
